// Encoder_48198122996375
// MI455X (gfx1250) — matrix operand order repaired and checked by reading, not compiled, not run on hardware
//
#include <hip/hip_runtime.h>
#include <hip/hip_bf16.h>
#include <stdint.h>

// ---------------------------------------------------------------------------
// CDNA5 (gfx1250) transformer encoder layer.
// All matmuls use v_wmma_f32_16x16x32_bf16 (bf16 in, fp32 accumulate).
// ---------------------------------------------------------------------------

typedef __attribute__((ext_vector_type(16))) __bf16 bf16x16;
typedef __attribute__((ext_vector_type(8)))  float  f32x8;

union FragU { bf16x16 v; uint32_t u[8]; };

__device__ __forceinline__ unsigned short f2bf(float x) {
    union { float f; uint32_t u; } c; c.f = x;
    uint32_t u = c.u;
    return (unsigned short)((u + 0x7fffu + ((u >> 16) & 1u)) >> 16);
}

__device__ __forceinline__ f32x8 wmma_bf16(bf16x16 a, bf16x16 b, f32x8 c) {
    // (neg_a, A, neg_b, B, c_mod, C, reuse_a, reuse_b)
    return __builtin_amdgcn_wmma_f32_16x16x32_bf16(false, a, false, b,
                                                   (short)0, c, false, false);
}

// A fragment (16x32, 16-bit): lane L holds row m=L&15, half h=L>>4.
// VGPR p (p<4): K = 2p..2p+1 (+8h);  p>=4: K = 2p+8..2p+9 (+8h).
__device__ __forceinline__ bf16x16 frag_a16(const unsigned short* row, int h) {
    FragU f;
#pragma unroll
    for (int p = 0; p < 8; ++p) {
        int k = (p < 4) ? (2 * p + 8 * h) : (2 * p + 8 + 8 * h);
        f.u[p] = (uint32_t)row[k] | ((uint32_t)row[k + 1] << 16);
    }
    return f.v;
}

// B fragment (32x16, 16-bit): lane L holds col n=L&15, half h=L>>4.
// VGPR p (p<4): K = 2p..2p+1 (+8h);  p>=4: K = 2p+8..2p+9 (+8h).  `row` points at LDS [n][k] row.
__device__ __forceinline__ bf16x16 frag_b16(const unsigned short* row, int h) {
    FragU f;
#pragma unroll
    for (int p = 0; p < 8; ++p) {
        int k = (p < 4) ? (2 * p + 8 * h) : (2 * p + 8 + 8 * h);
        f.u[p] = (uint32_t)row[k] | ((uint32_t)row[k + 1] << 16);
    }
    return f.v;
}

// ---------------------------------------------------------------------------
// GEMM: C[M,N] = A[M,K](bf16) @ B[K,N](bf16) + bias, with epilogue MODE:
//   1: out bf16 = (acc+bias)*scale            (QKV projections)
//   2: out f32  = acc+bias+res                (Wo / FFN2 residual adds)
//   3: out bf16 = gelu(acc+bias)              (FFN1)
// Block: 256 thr = 8 waves (4 M x 2 N), tile 128x128, k-step 32.
// ---------------------------------------------------------------------------
template <int MODE>
__global__ __launch_bounds__(256)
void gemm_bf16_kernel(const unsigned short* __restrict__ A,
                      const unsigned short* __restrict__ Bm,
                      const float* __restrict__ bias,
                      const float* __restrict__ res,
                      void* __restrict__ outp,
                      int N, int K, float scale) {
    __shared__ unsigned short Asm[128][40];   // [m][k] 32 + 8 pad
    __shared__ unsigned short Bsm[128][40];   // [n][k] transposed

    const int tid  = threadIdx.x;
    const int lane = tid & 31, wave = tid >> 5;
    const int wm = wave & 3, wn = wave >> 2;
    const int l15 = lane & 15, hh = lane >> 4;
    const int m0 = blockIdx.y * 128, n0 = blockIdx.x * 128;

    f32x8 zero8 = {0.f, 0.f, 0.f, 0.f, 0.f, 0.f, 0.f, 0.f};
    f32x8 acc[2][4];
#pragma unroll
    for (int i = 0; i < 2; ++i)
#pragma unroll
        for (int j = 0; j < 4; ++j) acc[i][j] = zero8;

    const int nk = K >> 5;
    for (int kt = 0; kt < nk; ++kt) {
        const int k0 = kt << 5;
        __syncthreads();
        // Stage A tile 128x32 (row-major, uint4 = 8 bf16 per load)
#pragma unroll
        for (int pass = 0; pass < 2; ++pass) {
            int c = tid + pass * 256;            // 0..511
            int row = c >> 2;                    // 4 chunks per row
            int kc  = (c & 3) << 3;
            uint4 d = *reinterpret_cast<const uint4*>(
                A + (size_t)(m0 + row) * K + k0 + kc);
            *reinterpret_cast<uint4*>(&Asm[row][kc]) = d;
        }
        // Stage B tile 32x128, transposed into [n][k]
#pragma unroll
        for (int pass = 0; pass < 2; ++pass) {
            int c = tid + pass * 256;            // 0..511
            int kk = c >> 4;                     // 16 chunks per k-row
            int nc = (c & 15) << 3;
            uint4 d = *reinterpret_cast<const uint4*>(
                Bm + (size_t)(k0 + kk) * N + n0 + nc);
            const unsigned short* ds = reinterpret_cast<const unsigned short*>(&d);
#pragma unroll
            for (int j = 0; j < 8; ++j) Bsm[nc + j][kk] = ds[j];
        }
        if (kt + 1 < nk) {  // gfx1250 global_prefetch_b8 for next A tile
            __builtin_prefetch(A + (size_t)(m0 + (tid >> 1)) * K + k0 + 32, 0, 1);
        }
        __syncthreads();

        bf16x16 af[2], bfr[4];
#pragma unroll
        for (int tm = 0; tm < 2; ++tm)
            af[tm] = frag_a16(&Asm[wm * 32 + tm * 16 + l15][0], hh);
#pragma unroll
        for (int tn = 0; tn < 4; ++tn)
            bfr[tn] = frag_b16(&Bsm[wn * 64 + tn * 16 + l15][0], hh);
#pragma unroll
        for (int tm = 0; tm < 2; ++tm)
#pragma unroll
            for (int tn = 0; tn < 4; ++tn)
                acc[tm][tn] = wmma_bf16(af[tm], bfr[tn], acc[tm][tn]);
    }

    // Epilogue. C layout: lane L -> col n=L&15, rows r+8*(L>>4) in VGPR r.
#pragma unroll
    for (int tm = 0; tm < 2; ++tm)
#pragma unroll
        for (int tn = 0; tn < 4; ++tn)
#pragma unroll
            for (int r = 0; r < 8; ++r) {
                int row = m0 + wm * 32 + tm * 16 + r + 8 * hh;
                int col = n0 + wn * 64 + tn * 16 + l15;
                float v = acc[tm][tn][r] + bias[col];
                size_t o = (size_t)row * N + col;
                if (MODE == 1) {
                    ((unsigned short*)outp)[o] = f2bf(v * scale);
                } else if (MODE == 2) {
                    ((float*)outp)[o] = v + res[o];
                } else {
                    float g = 0.5f * v * (1.f + erff(v * 0.70710678118f));
                    ((unsigned short*)outp)[o] = f2bf(g);
                }
            }
}

// ---------------------------------------------------------------------------
// Flash attention per (b, head, 128-query block).  q/k/v/o: bf16 [B,S,H*64].
// q is pre-scaled by 1/sqrt(64).  Key blocks of 64.  8 waves, 16 q-rows each.
// ---------------------------------------------------------------------------
__global__ __launch_bounds__(256)
void attention_kernel(const unsigned short* __restrict__ q,
                      const unsigned short* __restrict__ k,
                      const unsigned short* __restrict__ v,
                      const int* __restrict__ mask,
                      unsigned short* __restrict__ o,
                      int S) {
    __shared__ unsigned short Qs[128][72];      // [q][hd]
    __shared__ unsigned short Ks[64][72];       // [key][hd]  == B [n][k]
    __shared__ unsigned short Vt[64][72];       // [hd][key]  == B [n][k]
    __shared__ unsigned short Ps[8][16][72];    // per-wave P [row][key]
    __shared__ float maskAdd[64];

    const int tid  = threadIdx.x;
    const int lane = tid & 31, wave = tid >> 5;
    const int l15 = lane & 15, hh = lane >> 4;
    const int qb = blockIdx.x, head = blockIdx.y, b = blockIdx.z;
    const size_t rowbase = (size_t)b * S * 1024;
    const int cb = head * 64;

    // Stage Q tile (128 x 64)
#pragma unroll
    for (int pass = 0; pass < 4; ++pass) {
        int c = tid + pass * 256;                // 0..1023
        int r = c >> 3, hc = (c & 7) << 3;
        uint4 d = *reinterpret_cast<const uint4*>(
            q + rowbase + (size_t)(qb * 128 + r) * 1024 + cb + hc);
        *reinterpret_cast<uint4*>(&Qs[r][hc]) = d;
    }
    __syncthreads();

    bf16x16 aQ[2];
#pragma unroll
    for (int ks = 0; ks < 2; ++ks)
        aQ[ks] = frag_a16(&Qs[wave * 16 + l15][ks * 32], hh);

    f32x8 zero8 = {0.f, 0.f, 0.f, 0.f, 0.f, 0.f, 0.f, 0.f};
    f32x8 Oacc[4];
#pragma unroll
    for (int i = 0; i < 4; ++i) Oacc[i] = zero8;
    float mi[8], li[8];
#pragma unroll
    for (int r = 0; r < 8; ++r) { mi[r] = -3e38f; li[r] = 0.f; }

    const int nkb = S >> 6;
    for (int kb = 0; kb < nkb; ++kb) {
        __syncthreads();
        // Stage K (natural) and V (transposed): 64x64 each
#pragma unroll
        for (int pass = 0; pass < 2; ++pass) {
            int c = tid + pass * 256;            // 0..511
            int r = c >> 3, hc = (c & 7) << 3;
            const unsigned short* kp =
                k + rowbase + (size_t)(kb * 64 + r) * 1024 + cb + hc;
            uint4 dk = *reinterpret_cast<const uint4*>(kp);
            *reinterpret_cast<uint4*>(&Ks[r][hc]) = dk;
            const unsigned short* vp =
                v + rowbase + (size_t)(kb * 64 + r) * 1024 + cb + hc;
            uint4 dv = *reinterpret_cast<const uint4*>(vp);
            const unsigned short* dvs = reinterpret_cast<const unsigned short*>(&dv);
#pragma unroll
            for (int j = 0; j < 8; ++j) Vt[hc + j][r] = dvs[j];
        }
        if (tid < 64)
            maskAdd[tid] = (mask[b * S + kb * 64 + tid] == 0) ? -1e9f : 0.f;
        __syncthreads();

        // Scores S = Qb @ Kb^T  (4 tiles of 16 key columns)
        f32x8 sacc[4];
#pragma unroll
        for (int t = 0; t < 4; ++t) sacc[t] = zero8;
#pragma unroll
        for (int ks = 0; ks < 2; ++ks)
#pragma unroll
            for (int t = 0; t < 4; ++t) {
                bf16x16 bk = frag_b16(&Ks[t * 16 + l15][ks * 32], hh);
                sacc[t] = wmma_bf16(aQ[ks], bk, sacc[t]);
            }
        // Mask (additive, post-scale, matching reference)
#pragma unroll
        for (int t = 0; t < 4; ++t) {
            float ma = maskAdd[t * 16 + l15];
#pragma unroll
            for (int r = 0; r < 8; ++r) sacc[t][r] += ma;
        }
        // Row max (reduce over 16 lanes of each half-wave)
        float alpha[8], rs[8];
#pragma unroll
        for (int r = 0; r < 8; ++r) {
            float m = sacc[0][r];
#pragma unroll
            for (int t = 1; t < 4; ++t) m = fmaxf(m, sacc[t][r]);
            m = fmaxf(m, __shfl_xor(m, 1, 32));
            m = fmaxf(m, __shfl_xor(m, 2, 32));
            m = fmaxf(m, __shfl_xor(m, 4, 32));
            m = fmaxf(m, __shfl_xor(m, 8, 32));
            float mnew = fmaxf(mi[r], m);
            alpha[r] = __expf(mi[r] - mnew);
            mi[r] = mnew;
            rs[r] = 0.f;
        }
        // P = exp(S - rowmax) -> per-wave LDS (bf16), accumulate row sums
#pragma unroll
        for (int t = 0; t < 4; ++t)
#pragma unroll
            for (int r = 0; r < 8; ++r) {
                float p = __expf(sacc[t][r] - mi[r]);
                rs[r] += p;
                Ps[wave][r + 8 * hh][t * 16 + l15] = f2bf(p);
            }
#pragma unroll
        for (int r = 0; r < 8; ++r) {
            float s = rs[r];
            s += __shfl_xor(s, 1, 32);
            s += __shfl_xor(s, 2, 32);
            s += __shfl_xor(s, 4, 32);
            s += __shfl_xor(s, 8, 32);
            li[r] = li[r] * alpha[r] + s;
        }
        // Rescale running O, then O += P @ V (same-wave LDS is in-order)
#pragma unroll
        for (int tn = 0; tn < 4; ++tn)
#pragma unroll
            for (int r = 0; r < 8; ++r) Oacc[tn][r] *= alpha[r];
#pragma unroll
        for (int ks2 = 0; ks2 < 2; ++ks2) {
            bf16x16 aP = frag_a16(&Ps[wave][l15][ks2 * 32], hh);
#pragma unroll
            for (int tn = 0; tn < 4; ++tn) {
                bf16x16 bv = frag_b16(&Vt[tn * 16 + l15][ks2 * 32], hh);
                Oacc[tn] = wmma_bf16(aP, bv, Oacc[tn]);
            }
        }
    }
    // Normalize and store (concat-head layout [b][s][h*64+hd])
    float inv[8];
#pragma unroll
    for (int r = 0; r < 8; ++r) inv[r] = 1.f / li[r];
#pragma unroll
    for (int tn = 0; tn < 4; ++tn)
#pragma unroll
        for (int r = 0; r < 8; ++r) {
            int row = qb * 128 + wave * 16 + r + 8 * hh;
            int col = cb + tn * 16 + l15;
            o[rowbase + (size_t)row * 1024 + col] = f2bf(Oacc[tn][r] * inv[r]);
        }
}

// ---------------------------------------------------------------------------
// LayerNorm over D=1024, fp32 in -> bf16 out. One 256-thread block per row.
// ---------------------------------------------------------------------------
__global__ __launch_bounds__(256)
void layernorm_kernel(const float* __restrict__ x, const float* __restrict__ g,
                      const float* __restrict__ bet, unsigned short* __restrict__ out) {
    __shared__ float r1[256], r2[256];
    const int row = blockIdx.x, tid = threadIdx.x;
    const float* xr = x + (size_t)row * 1024;
    float4 xv = *reinterpret_cast<const float4*>(xr + tid * 4);
    r1[tid] = xv.x + xv.y + xv.z + xv.w;
    r2[tid] = xv.x * xv.x + xv.y * xv.y + xv.z * xv.z + xv.w * xv.w;
    __syncthreads();
    for (int off = 128; off > 0; off >>= 1) {
        if (tid < off) { r1[tid] += r1[tid + off]; r2[tid] += r2[tid + off]; }
        __syncthreads();
    }
    float mu = r1[0] * (1.f / 1024.f);
    float var = r2[0] * (1.f / 1024.f) - mu * mu;
    float rstd = rsqrtf(var + 1e-6f);
    int c = tid * 4;
    float4 gv = *reinterpret_cast<const float4*>(g + c);
    float4 bv = *reinterpret_cast<const float4*>(bet + c);
    unsigned short* orow = out + (size_t)row * 1024 + c;
    orow[0] = f2bf((xv.x - mu) * rstd * gv.x + bv.x);
    orow[1] = f2bf((xv.y - mu) * rstd * gv.y + bv.y);
    orow[2] = f2bf((xv.z - mu) * rstd * gv.z + bv.z);
    orow[3] = f2bf((xv.w - mu) * rstd * gv.w + bv.w);
}

// ---------------------------------------------------------------------------
// Weight conversion: fp32 -> bf16.  QKV variant reorders [H,D,HD]->[D,H*HD].
// ---------------------------------------------------------------------------
__global__ __launch_bounds__(256)
void cvt_qkv_w_kernel(const float* __restrict__ W, unsigned short* __restrict__ out) {
    int i = blockIdx.x * 256 + threadIdx.x;       // i = h*65536 + d*64 + j
    int j = i & 63;
    int d = (i >> 6) & 1023;
    int h = i >> 16;
    out[(size_t)d * 1024 + h * 64 + j] = f2bf(W[i]);
}

__global__ __launch_bounds__(256)
void cvt_plain_kernel(const float* __restrict__ W, unsigned short* __restrict__ out, int n) {
    int i = blockIdx.x * 256 + threadIdx.x;
    if (i < n) out[i] = f2bf(W[i]);
}

// ---------------------------------------------------------------------------
extern "C" void kernel_launch(void* const* d_in, const int* in_sizes, int n_in,
                              void* d_out, int out_size, void* d_ws, size_t ws_size,
                              hipStream_t stream) {
    (void)in_sizes; (void)n_in; (void)out_size; (void)ws_size;
    const float* hidden = (const float*)d_in[0];
    const int*   mask   = (const int*)d_in[1];
    const float* Wq = (const float*)d_in[2];  const float* bq = (const float*)d_in[3];
    const float* Wk = (const float*)d_in[4];  const float* bk = (const float*)d_in[5];
    const float* Wv = (const float*)d_in[6];  const float* bv = (const float*)d_in[7];
    const float* Wo = (const float*)d_in[8];  const float* bo = (const float*)d_in[9];
    const float* g1 = (const float*)d_in[10]; const float* be1 = (const float*)d_in[11];
    const float* g2 = (const float*)d_in[12]; const float* be2 = (const float*)d_in[13];
    const float* W1 = (const float*)d_in[14]; const float* b1 = (const float*)d_in[15];
    const float* W2 = (const float*)d_in[16]; const float* b2 = (const float*)d_in[17];

    constexpr size_t MB = 1u << 20;
    char* ws = (char*)d_ws;
    unsigned short* x1b  = (unsigned short*)(ws + 0 * MB);    // 8 MB
    unsigned short* wqb  = (unsigned short*)(ws + 8 * MB);    // 2 MB
    unsigned short* wkb  = (unsigned short*)(ws + 10 * MB);   // 2 MB
    unsigned short* wvb  = (unsigned short*)(ws + 12 * MB);   // 2 MB
    unsigned short* wob  = (unsigned short*)(ws + 14 * MB);   // 2 MB
    unsigned short* w1b  = (unsigned short*)(ws + 16 * MB);   // 8 MB
    unsigned short* w2b  = (unsigned short*)(ws + 24 * MB);   // 8 MB
    unsigned short* qb_  = (unsigned short*)(ws + 32 * MB);   // 8 MB
    unsigned short* kb_  = (unsigned short*)(ws + 40 * MB);   // 8 MB
    unsigned short* vb_  = (unsigned short*)(ws + 48 * MB);   // 8 MB
    unsigned short* attb = (unsigned short*)(ws + 56 * MB);   // 8 MB
    float*          hbuf = (float*)         (ws + 64 * MB);   // 16 MB
    unsigned short* x2b  = (unsigned short*)(ws + 80 * MB);   // 8 MB
    unsigned short* ff1b = (unsigned short*)(ws + 88 * MB);   // 32 MB

    const int M = 4096;   // B*S
    const int D = 1024, F = 4096, S = 1024;

    // Weight conversion (deterministic, every launch)
    cvt_qkv_w_kernel<<<4096, 256, 0, stream>>>(Wq, wqb);
    cvt_qkv_w_kernel<<<4096, 256, 0, stream>>>(Wk, wkb);
    cvt_qkv_w_kernel<<<4096, 256, 0, stream>>>(Wv, wvb);
    cvt_plain_kernel<<<4096, 256, 0, stream>>>(Wo, wob, D * D);
    cvt_plain_kernel<<<16384, 256, 0, stream>>>(W1, w1b, D * F);
    cvt_plain_kernel<<<16384, 256, 0, stream>>>(W2, w2b, F * D);

    // x1 = LN1(hidden)  (bf16)
    layernorm_kernel<<<M, 256, 0, stream>>>(hidden, g1, be1, x1b);

    // QKV projections (q pre-scaled by 1/sqrt(64))
    gemm_bf16_kernel<1><<<dim3(D / 128, M / 128), 256, 0, stream>>>(
        x1b, wqb, bq, nullptr, qb_, D, D, 0.125f);
    gemm_bf16_kernel<1><<<dim3(D / 128, M / 128), 256, 0, stream>>>(
        x1b, wkb, bk, nullptr, kb_, D, D, 1.f);
    gemm_bf16_kernel<1><<<dim3(D / 128, M / 128), 256, 0, stream>>>(
        x1b, wvb, bv, nullptr, vb_, D, D, 1.f);

    // Flash attention: grid (S/128 q-blocks, H heads, B batches)
    attention_kernel<<<dim3(8, 16, 4), 256, 0, stream>>>(qb_, kb_, vb_, mask, attb, S);

    // h = attn @ Wo + bo + hidden  (fp32)
    gemm_bf16_kernel<2><<<dim3(D / 128, M / 128), 256, 0, stream>>>(
        attb, wob, bo, hidden, hbuf, D, D, 1.f);

    // x2 = LN2(h)
    layernorm_kernel<<<M, 256, 0, stream>>>(hbuf, g2, be2, x2b);

    // ff1 = gelu(x2 @ W1 + b1)  (bf16)
    gemm_bf16_kernel<3><<<dim3(F / 128, M / 128), 256, 0, stream>>>(
        x2b, w1b, b1, nullptr, ff1b, F, D, 1.f);

    // out = ff1 @ W2 + b2 + h  (fp32)
    gemm_bf16_kernel<2><<<dim3(D / 128, M / 128), 256, 0, stream>>>(
        ff1b, w2b, b2, hbuf, (float*)d_out, D, F, 1.f);
}
